// UtoU_695784702049
// MI455X (gfx1250) — matrix operand order repaired and checked by reading, not compiled, not run on hardware
//
#include <hip/hip_runtime.h>
#include <hip/hip_bf16.h>

// N=20000 nodes, K=16 colors, C=O=256 channels.
// Main GEMM: (N*K,256)x(256,256) via v_wmma_f32_16x16x32_bf16, f32 accumulate.
// A 16x16 output tile == one node's 16 colors x 16 out-channels -> the epilogue
// (bias, scatter-add z2 at row coloring[n], masked z3, residual) fuses per-tile.

typedef __attribute__((ext_vector_type(16))) __bf16 v16bf;
typedef __attribute__((ext_vector_type(8)))  __bf16 v8bf;
typedef __attribute__((ext_vector_type(8)))  float  v8f;

#define NNODES 20000
#define KC 16
#define CH 256
#define LDSTR 260  // 260 % 64 == 4 -> 16 rows hit 16 distinct banks on column reads

// ---------------- prep: weights -> bf16 in WMMA B-layout ----------------
// B tile layout (wave32, 16x16x32 bf16): lane l<16 holds column o=l,
// K = {kbase+0..7, kbase+16..23}; lane l>=16 holds column o=l-16, K shifted by 8.
// Stored as [otile][ktile][lane][16] so compute does one direct v16bf load.
__global__ __launch_bounds__(256) void prep_weights_kernel(
    const float* __restrict__ W1, const float* __restrict__ W2,
    const float* __restrict__ W3, __bf16* __restrict__ w1bf,
    __bf16* __restrict__ w23bf) {
  int idx = blockIdx.x * 256 + threadIdx.x;
  if (idx < 16 * 8 * 32 * 16) {  // W1: 16 otiles x 8 ktiles x 32 lanes x 16 elems
    int e = idx & 15, lane = (idx >> 4) & 31, kt = (idx >> 9) & 7, ot = idx >> 12;
    int o = ot * 16 + (lane & 15);
    int k = kt * 32 + (lane >> 4) * 8 + (e & 7) + (e >> 3) * 16;
    w1bf[idx] = (__bf16)W1[o * CH + k];
  } else {
    int j = idx - 16 * 8 * 32 * 16;
    if (j >= 32 * 8 * 32 * 16) return;  // W23: 32 otiles (512 stacked cols)
    int e = j & 15, lane = (j >> 4) & 31, kt = (j >> 9) & 7, ot = j >> 12;
    int oc = ot * 16 + (lane & 15);
    int k = kt * 32 + (lane >> 4) * 8 + (e & 7) + (e >> 3) * 16;
    float w = (oc < CH) ? 0.1f * W2[oc * CH + k] : 0.1f * W3[(oc - CH) * CH + k];
    w23bf[j] = (__bf16)w;
  }
}

// ---------------- prep: mean2[n,c] = (sum_k u[n,k,c]) / n_batch[n] ----------
__global__ __launch_bounds__(256) void mean2_kernel(
    const float* __restrict__ u, const float* __restrict__ n_batch,
    __bf16* __restrict__ mean2) {
  int idx = blockIdx.x * 256 + threadIdx.x;  // n*256 + c
  if (idx >= NNODES * CH) return;
  int n = idx >> 8;
  const float* up = u + (size_t)n * KC * CH + (idx & 255);
  float s = 0.f;
#pragma unroll
  for (int k = 0; k < KC; ++k) s += up[k * CH];
  mean2[idx] = (__bf16)(s * (1.0f / n_batch[n]));
}

// ---------------- z-GEMM: Z[20000,512] = mean2 @ [0.1*W2 || 0.1*W3]^T -------
// One wave per 16x32 tile pair (ct, ct+16) sharing each A fragment.
// A layout (16-bit A 16x32): lane<16 holds row m=lane, K={kb..kb+7,kb+16..kb+23};
// lane>=16 holds row m=lane-16 with K shifted by 8.
__global__ __launch_bounds__(32) void zgemm_kernel(
    const __bf16* __restrict__ A, const __bf16* __restrict__ B,
    float* __restrict__ Z) {
  int rt = blockIdx.x, ct = blockIdx.y;  // ct in 0..15
  int lane = threadIdx.x, half = lane >> 4, m = lane & 15;
  int row = rt * 16 + m;
  v8f acc0 = {}, acc1 = {};
#pragma unroll
  for (int kt = 0; kt < 8; ++kt) {
    const __bf16* ap = A + (size_t)row * CH + kt * 32 + half * 8;
    v8bf a0 = *(const v8bf*)ap;
    v8bf a1 = *(const v8bf*)(ap + 16);
    v16bf a;
#pragma unroll
    for (int e = 0; e < 8; ++e) { a[e] = a0[e]; a[8 + e] = a1[e]; }
    v16bf b0 = *(const v16bf*)(B + (((size_t)ct * 8 + kt) * 32 + lane) * 16);
    v16bf b1 = *(const v16bf*)(B + ((((size_t)ct + 16) * 8 + kt) * 32 + lane) * 16);
    acc0 = __builtin_amdgcn_wmma_f32_16x16x32_bf16(false, a, false, b0,
                                                   (short)0, acc0, false, false);
    acc1 = __builtin_amdgcn_wmma_f32_16x16x32_bf16(false, a, false, b1,
                                                   (short)0, acc1, false, false);
  }
  // C/D layout: VGPR r -> row M = r + 8*half, col N = lane&15
#pragma unroll
  for (int r = 0; r < 8; ++r) {
    size_t zr = (size_t)(rt * 16 + r + 8 * half) * 512;
    Z[zr + ct * 16 + m] = acc0[r];
    Z[zr + (ct + 16) * 16 + m] = acc1[r];
  }
}

// ---------------- main fused kernel: one block (8 waves) per node -----------
__global__ __launch_bounds__(256) void node_gemm_kernel(
    const float* __restrict__ u, const __bf16* __restrict__ w1bf,
    const float* __restrict__ zbuf, const float* __restrict__ b1,
    const unsigned char* __restrict__ mask, const int* __restrict__ coloring,
    float* __restrict__ out) {
  __shared__ float su[KC * LDSTR];                 // f32 tile (residual source)
  __shared__ __align__(32) __bf16 sa[8 * 32 * 16]; // bf16 A fragments, WMMA layout
  int n = blockIdx.x;
  int t = threadIdx.x;
  const float* un = u + (size_t)n * KC * CH;

  // Stage u[n] (16 KB) into padded LDS with async DMA (ASYNCcnt-tracked).
  // LDS flat addresses truncate to the LDS byte offset (ISA 10.2 aperture map).
#pragma unroll
  for (int i = 0; i < 4; ++i) {
    int g = t + i * 256;                       // 16-byte chunk id, 1024 total
    int row = g >> 6, col = (g & 63) * 4;      // chunk stays within one row
    unsigned lds = (unsigned)(uintptr_t)&su[row * LDSTR + col];
    unsigned long long ga = (unsigned long long)(uintptr_t)(un + g * 4);
    asm volatile("global_load_async_to_lds_b128 %0, %1, off"
                 :: "v"(lds), "v"(ga) : "memory");
  }
  asm volatile("s_wait_asynccnt 0" ::: "memory");
  __syncthreads();

  int kt0 = t >> 5, lane = t & 31, half = lane >> 4, m = lane & 15;

  // Convert once: each thread builds exactly one (kt,lane) A fragment (32 B).
  {
    const float* ap = &su[m * LDSTR + kt0 * 32 + half * 8];
    v16bf a;
#pragma unroll
    for (int e = 0; e < 8; ++e) {
      a[e] = (__bf16)ap[e];
      a[8 + e] = (__bf16)ap[16 + e];
    }
    *(v16bf*)(sa + t * 16) = a;
  }
  __syncthreads();

  int wave = t >> 5;
  int ot0 = wave, ot1 = wave + 8;
  int color = coloring[n];

  v8f acc0 = {}, acc1 = {};
#pragma unroll
  for (int kt = 0; kt < 8; ++kt) {
    v16bf a = *(const v16bf*)(sa + ((kt * 32 + lane) << 4));
    v16bf b0 = *(const v16bf*)(w1bf + (((size_t)ot0 * 8 + kt) * 32 + lane) * 16);
    v16bf b1 = *(const v16bf*)(w1bf + (((size_t)ot1 * 8 + kt) * 32 + lane) * 16);
    acc0 = __builtin_amdgcn_wmma_f32_16x16x32_bf16(false, a, false, b0,
                                                   (short)0, acc0, false, false);
    acc1 = __builtin_amdgcn_wmma_f32_16x16x32_bf16(false, a, false, b1,
                                                   (short)0, acc1, false, false);
  }

  // Fused epilogue: residual + b1 + scatter(z2 at k==color) + mask*z3.
#pragma unroll
  for (int oi = 0; oi < 2; ++oi) {
    int o = (oi ? ot1 : ot0) * 16 + m;
    float bias = b1[o];
    float z2v = zbuf[(size_t)n * 512 + o];       // 0.1*(mean2 @ W2^T)
    float z3v = zbuf[(size_t)n * 512 + CH + o];  // 0.1*(mean2 @ W3^T)
    v8f acc = oi ? acc1 : acc0;
#pragma unroll
    for (int r = 0; r < 8; ++r) {
      int k = r + 8 * half;  // color row of this accumulator element
      float v = su[k * LDSTR + o] + acc[r] + bias;
      if (color == k) v += z2v;
      if (mask[n * KC + k]) v += z3v;
      out[((size_t)n * KC + k) * CH + o] = v;
    }
  }
}

// ---------------- launch ----------------------------------------------------
extern "C" void kernel_launch(void* const* d_in, const int* in_sizes, int n_in,
                              void* d_out, int out_size, void* d_ws, size_t ws_size,
                              hipStream_t stream) {
  const float* u        = (const float*)d_in[0];
  const float* n_batch  = (const float*)d_in[1];
  const float* W1       = (const float*)d_in[2];
  const float* b1       = (const float*)d_in[3];
  const float* W2       = (const float*)d_in[4];
  const float* W3       = (const float*)d_in[5];
  const unsigned char* mask = (const unsigned char*)d_in[6];  // jnp bool = 1B
  const int* coloring   = (const int*)d_in[7];
  float* out            = (float*)d_out;

  char* ws = (char*)d_ws;
  __bf16* w1bf  = (__bf16*)(ws);             // 131,072 B
  __bf16* w23bf = (__bf16*)(ws + 131072);    // 262,144 B
  __bf16* mean2 = (__bf16*)(ws + 393216);    // 10,240,000 B
  float*  zbuf  = (float*) (ws + 10633216);  // 40,960,000 B (~49 MB total)

  prep_weights_kernel<<<(16 * 8 * 32 * 16 + 32 * 8 * 32 * 16 + 255) / 256, 256, 0,
                        stream>>>(W1, W2, W3, w1bf, w23bf);
  mean2_kernel<<<(NNODES * CH + 255) / 256, 256, 0, stream>>>(u, n_batch, mean2);

  // Z = mean2 @ [0.1W2 || 0.1W3]^T  (all operands L2-resident)
  zgemm_kernel<<<dim3(NNODES / 16, 16), 32, 0, stream>>>(mean2, w23bf, zbuf);

  // Fused lin1 GEMM + epilogue, one block per node.
  node_gemm_kernel<<<NNODES, 256, 0, stream>>>(u, w1bf, zbuf, b1, mask, coloring,
                                               out);
}
